// _NLAttentionModule_77163382439958
// MI455X (gfx1250) — matrix operand order repaired and checked by reading, not compiled, not run on hardware
//
#include <hip/hip_runtime.h>
#include <hip/hip_bf16.h>
#include <stdint.h>

// ---------------------------------------------------------------------------
// Non-local attention (embedded gaussian, softmax over QUERY axis) for MI455X.
// All GEMMs via v_wmma_f32_16x16x32_bf16 (f32 accumulate). Workspace ~33 MB.
//
// Pipeline (6 launches, S computed twice):
//   proj_gemm x3  : q/k/v channel projections -> bf16 (vp stored transposed)
//   col_stats     : online-softmax column stats (max,sum) over the query axis
//   attn_apply    : recompute S tile -> exp/scale -> LDS -> chain WMMA vs vpT
//   out_proj      : out @ W_out^T + v (fused residual)
// ---------------------------------------------------------------------------

typedef __attribute__((ext_vector_type(16))) __bf16 v16bf;
typedef __attribute__((ext_vector_type(8)))  float  v8f;

namespace {
constexpr int   kB = 8, kC = 256, kD = 128, kN = 64 * 64;       // N = 4096
constexpr float kScale = 0.08838834764831845f;                  // 1/sqrt(128)
}

union BFrag {
  v16bf          v;
  uint4          q[2];
  unsigned short s[16];
};

static __device__ __forceinline__ unsigned short f2bf(float f) {
  unsigned u = __float_as_uint(f);
  u += 0x7FFFu + ((u >> 16) & 1u);      // round-to-nearest-even
  return (unsigned short)(u >> 16);
}

static __device__ __forceinline__ v8f wmma_bf16(v16bf a, v16bf b, v8f c) {
  return __builtin_amdgcn_wmma_f32_16x16x32_bf16(false, a, false, b,
                                                 (short)0, c, false, false);
}

// A fragment: 16x32 bf16, src row-major [*, lda]. lane&15 = row M,
// K-run: kbase + half*16 .. +15 (32 contiguous bytes).
static __device__ __forceinline__ v16bf load_a(const unsigned short* src,
                                               int row, int lda, int kbase, int half) {
  BFrag f;
  const unsigned short* p = src + (size_t)row * lda + kbase + half * 16;
  f.q[0] = *(const uint4*)(p);
  f.q[1] = *(const uint4*)(p + 8);
  return f.v;
}

// B fragment: 32x16 bf16. src is B^T row-major: src[ncol][k], lane&15 = col N,
// K-run: kbase + half*16 .. +15 (32 contiguous bytes).
static __device__ __forceinline__ v16bf load_b(const unsigned short* src,
                                               int ncol, int ldb, int kbase, int half) {
  BFrag f;
  const unsigned short* p = src + (size_t)ncol * ldb + kbase + half * 16;
  f.q[0] = *(const uint4*)(p);
  f.q[1] = *(const uint4*)(p + 8);
  return f.v;
}

// Same as load_b but source is f32 (weights), converted to bf16 on the fly.
static __device__ __forceinline__ v16bf load_b_f32(const float* src,
                                                   int ncol, int ldb, int kbase, int half) {
  BFrag f;
  const float* p = src + (size_t)ncol * ldb + kbase + half * 16;
#pragma unroll
  for (int i = 0; i < 16; ++i) f.s[i] = f2bf(p[i]);
  return f.v;
}

// ---------------------------------------------------------------------------
// Kernel 1: channel projection. out[n][d] = sum_c x[b][c][n] * W[d][c]
// STORE_T=false -> out row-major [B][N][D]; STORE_T=true -> out [B][D][N].
// Workgroup: 4 waves, tile 16n x 64d. grid (N/16, D/64, B).
// ---------------------------------------------------------------------------
template <bool STORE_T>
__global__ __launch_bounds__(128) void proj_gemm(const float* __restrict__ x,
                                                 const float* __restrict__ w,
                                                 unsigned short* __restrict__ out) {
  __shared__ __align__(16) unsigned short ldsXT[16][32];   // x^T tile, bf16
  __shared__ __align__(16) unsigned short ldsO[16][64];    // output transpose staging
  const int t = threadIdx.x;
  const int wave = t >> 5, lane = t & 31, half = lane >> 4, col = lane & 15;
  const int b = blockIdx.z;
  const int n0 = blockIdx.x * 16;
  const int dgrp = blockIdx.y * 64;
  const int d0 = dgrp + wave * 16;
  const float* xb = x + (size_t)b * kC * kN;

  v8f acc = {};
  for (int c0 = 0; c0 < kC; c0 += 32) {
    __syncthreads();
#pragma unroll
    for (int i = 0; i < 4; ++i) {               // 32c x 16n tile, coalesced in n
      int idx = t + i * 128;
      int cl = idx >> 4, nl = idx & 15;
      ldsXT[nl][cl] = f2bf(xb[(size_t)(c0 + cl) * kN + n0 + nl]);
    }
    __syncthreads();
    v16bf a  = load_a(&ldsXT[0][0], col, 32, 0, half);
    v16bf bb = load_b_f32(w, d0 + col, kC, c0, half);     // W[d][c], c contiguous
    acc = wmma_bf16(a, bb, acc);
  }

  if (STORE_T) {
    // lane owns column d = d0+col, rows n = n0 + half*8 + r (contiguous!)
    BFrag pk;
#pragma unroll
    for (int r = 0; r < 8; ++r) pk.s[r] = f2bf(acc[r]);
    unsigned short* dst = out + ((size_t)b * kD + d0 + col) * kN + n0 + half * 8;
    *(uint4*)dst = pk.q[0];
  } else {
    __syncthreads();
#pragma unroll
    for (int r = 0; r < 8; ++r) ldsO[half * 8 + r][wave * 16 + col] = f2bf(acc[r]);
    __syncthreads();
    int n = t >> 3, seg = t & 7;                // cooperative 16B row stores
    uint4 val = *(uint4*)&ldsO[n][seg * 8];
    unsigned short* dst = out + ((size_t)b * kN + n0 + n) * kD + dgrp + seg * 8;
    *(uint4*)dst = val;
  }
}

// ---------------------------------------------------------------------------
// Kernel 2: fused column stats of S = qp kp^T * scale, softmax over n per m.
// Per-lane ONLINE softmax (column m is fixed per lane): max+sum in ONE S-pass.
// All 4 A-fragment loads issued BEFORE the WMMA chain (keeps loads in flight
// under the matrix pipe); exp terms tree-summed to avoid serial TRANS stalls.
// grid (N/16, 1, B), 4 waves; wave strides over n-tiles. No atomics needed.
// ---------------------------------------------------------------------------
__global__ __launch_bounds__(128) void col_stats(const unsigned short* __restrict__ qp,
                                                 const unsigned short* __restrict__ kp,
                                                 float* __restrict__ cmax,
                                                 float* __restrict__ csum) {
  __shared__ float sMax[4][16];
  __shared__ float sSum[4][16];
  const int t = threadIdx.x, wave = t >> 5, lane = t & 31;
  const int half = lane >> 4, col = lane & 15;
  const int b = blockIdx.z, m0 = blockIdx.x * 16;
  const unsigned short* qb = qp + (size_t)b * kN * kD;
  const unsigned short* kb = kp + (size_t)b * kN * kD;

  v16bf bk[4];                                   // kp tile: invariant over n loop
#pragma unroll
  for (int ks = 0; ks < 4; ++ks) bk[ks] = load_b(kb, m0 + col, kD, ks * 32, half);

  float runmax = -3.4e38f, runsum = 0.0f;
  for (int nt = wave; nt < kN / 16; nt += 4) {
    __builtin_prefetch(qb + (size_t)((nt + 4) * 16 + col) * kD, 0, 1);

    v16bf a[4];                                  // hoist: all loads before WMMAs
#pragma unroll
    for (int ks = 0; ks < 4; ++ks) a[ks] = load_a(qb, nt * 16 + col, kD, ks * 32, half);
    v8f s = {};
#pragma unroll
    for (int ks = 0; ks < 4; ++ks) s = wmma_bf16(a[ks], bk[ks], s);

    // online max/sum update (tree-structured to pipeline the exp unit)
    float m01 = fmaxf(s[0], s[1]), m23 = fmaxf(s[2], s[3]);
    float m45 = fmaxf(s[4], s[5]), m67 = fmaxf(s[6], s[7]);
    float tmax = fmaxf(fmaxf(m01, m23), fmaxf(m45, m67)) * kScale;
    const float M = fmaxf(runmax, tmax);
    float e[8];
#pragma unroll
    for (int r = 0; r < 8; ++r) e[r] = __expf(s[r] * kScale - M);
    const float rs = __expf(runmax - M);
    const float sum = ((e[0] + e[1]) + (e[2] + e[3])) + ((e[4] + e[5]) + (e[6] + e[7]));
    runsum = runsum * rs + sum;
    runmax = M;
  }

  // merge lane halves (rows 0-7 vs 8-15 of each tile)
  {
    const float om = __shfl_xor(runmax, 16, 32);
    const float os = __shfl_xor(runsum, 16, 32);
    const float M  = fmaxf(runmax, om);
    runsum = runsum * __expf(runmax - M) + os * __expf(om - M);
    runmax = M;
  }
  if (half == 0) { sMax[wave][col] = runmax; sSum[wave][col] = runsum; }
  __syncthreads();

  if (t < 16) {                                  // merge the 4 waves, final write
    float M = sMax[0][t];
#pragma unroll
    for (int wv = 1; wv < 4; ++wv) M = fmaxf(M, sMax[wv][t]);
    float S = 0.0f;
#pragma unroll
    for (int wv = 0; wv < 4; ++wv) S += sSum[wv][t] * __expf(sMax[wv][t] - M);
    cmax[(size_t)b * kN + m0 + t] = M;
    csum[(size_t)b * kN + m0 + t] = S;
  }
}

// ---------------------------------------------------------------------------
// Kernel 3: out[n][d] = sum_m exp(S[n,m]*scale - cmax[m])/csum[m] * vp[m][d].
// grid (N/16, 1, B), 4 waves. Per 64-wide m step: each wave builds one 16x16
// P subtile (WMMA chain K=128), writes bf16 P to LDS, then all waves chain
// P(16x64) x vpT into 2 persistent 16x16 accumulators (wave covers 32 d).
// ---------------------------------------------------------------------------
__global__ __launch_bounds__(128) void attn_apply(const unsigned short* __restrict__ qp,
                                                  const unsigned short* __restrict__ kp,
                                                  const unsigned short* __restrict__ vpT,
                                                  const float* __restrict__ cmax,
                                                  const float* __restrict__ csum,
                                                  unsigned short* __restrict__ outb) {
  __shared__ __align__(16) unsigned short ldsP[16][64];    // P tile (n x m), bf16
  __shared__ __align__(16) unsigned short ldsO[16][128];   // out transpose staging
  const int t = threadIdx.x, wave = t >> 5, lane = t & 31;
  const int half = lane >> 4, col = lane & 15;
  const int b = blockIdx.z, n0 = blockIdx.x * 16;
  const unsigned short* qb = qp + (size_t)b * kN * kD;
  const unsigned short* kb = kp + (size_t)b * kN * kD;
  const unsigned short* vb = vpT + (size_t)b * kD * kN;
  const size_t bN = (size_t)b * kN;
  const int d_base = wave * 32;

  v16bf aq[4];                                   // qp rows: invariant over m loop
#pragma unroll
  for (int ks = 0; ks < 4; ++ks) aq[ks] = load_a(qb, n0 + col, kD, ks * 32, half);

  v8f acc0 = {}, acc1 = {};
  for (int mbase = 0; mbase < kN; mbase += 64) {
    const int msub = mbase + wave * 16;
    if (mbase + 64 < kN) {                       // stay one m-tile ahead in L2/L0
      __builtin_prefetch(kb + (size_t)(msub + 64 + col) * kD, 0, 1);
      __builtin_prefetch(vb + (size_t)(d_base + col) * kN + mbase + 64, 0, 1);
      __builtin_prefetch(vb + (size_t)(d_base + 16 + col) * kN + mbase + 64, 0, 1);
    }

    // ---- stage 1: this wave's 16x16 score subtile (m = msub..msub+15) ----
    v16bf bkf[4];                                // hoist: all loads before WMMAs
#pragma unroll
    for (int ks = 0; ks < 4; ++ks) bkf[ks] = load_b(kb, msub + col, kD, ks * 32, half);
    v8f s = {};
#pragma unroll
    for (int ks = 0; ks < 4; ++ks) s = wmma_bf16(aq[ks], bkf[ks], s);

    const int m = msub + col;
    const float cm = cmax[bN + m];
    const float ci = 1.0f / csum[bN + m];
    float e[8];
#pragma unroll
    for (int r = 0; r < 8; ++r) e[r] = __expf(s[r] * kScale - cm);  // independent
#pragma unroll
    for (int r = 0; r < 8; ++r)
      ldsP[half * 8 + r][wave * 16 + col] = f2bf(e[r] * ci);
    __syncthreads();

    // ---- stage 2: P(16x64) @ vp(64x32 slice per wave) ----
    v16bf bv[4];                                 // hoist global vp loads
#pragma unroll
    for (int ks2 = 0; ks2 < 2; ++ks2) {
      bv[2 * ks2]     = load_b(vb, d_base + col,      kN, mbase + ks2 * 32, half);
      bv[2 * ks2 + 1] = load_b(vb, d_base + 16 + col, kN, mbase + ks2 * 32, half);
    }
#pragma unroll
    for (int ks2 = 0; ks2 < 2; ++ks2) {
      v16bf ap = load_a(&ldsP[0][0], col, 64, ks2 * 32, half);
      acc0 = wmma_bf16(ap, bv[2 * ks2],     acc0);
      acc1 = wmma_bf16(ap, bv[2 * ks2 + 1], acc1);
    }
    __syncthreads();                             // protect ldsP for next iter
  }

  // epilogue: transpose via LDS, store row-major bf16 [B][N][D]
#pragma unroll
  for (int r = 0; r < 8; ++r) {
    ldsO[half * 8 + r][d_base + col]      = f2bf(acc0[r]);
    ldsO[half * 8 + r][d_base + 16 + col] = f2bf(acc1[r]);
  }
  __syncthreads();
  int n = t >> 3, seg = t & 7;                   // 32B per thread
  uint4 v0 = *(uint4*)&ldsO[n][seg * 16];
  uint4 v1 = *(uint4*)&ldsO[n][seg * 16 + 8];
  unsigned short* dst = outb + (bN + n0 + n) * kD + seg * 16;
  *(uint4*)(dst)     = v0;
  *(uint4*)(dst + 8) = v1;
}

// ---------------------------------------------------------------------------
// Kernel 4: y[b][c][n] = v[b][c][n] + sum_d out[n][d] * W_out[c][d]
// grid (N/16, C/64, B), 4 waves (wave -> 16 c columns).
// ---------------------------------------------------------------------------
__global__ __launch_bounds__(128) void out_proj(const unsigned short* __restrict__ outb,
                                                const float* __restrict__ wout,
                                                const float* __restrict__ vin,
                                                float* __restrict__ y) {
  const int t = threadIdx.x, wave = t >> 5, lane = t & 31;
  const int half = lane >> 4, col = lane & 15;
  const int b = blockIdx.z, n0 = blockIdx.x * 16;
  const int c0 = blockIdx.y * 64 + wave * 16;
  const unsigned short* ob = outb + (size_t)b * kN * kD;

  v16bf a[4], bb[4];                             // hoist loads before WMMAs
#pragma unroll
  for (int ks = 0; ks < 4; ++ks) {
    a[ks]  = load_a(ob, n0 + col, kD, ks * 32, half);
    bb[ks] = load_b_f32(wout, c0 + col, kD, ks * 32, half);    // W_out[c][d]
  }
  v8f acc = {};
#pragma unroll
  for (int ks = 0; ks < 4; ++ks) acc = wmma_bf16(a[ks], bb[ks], acc);

  // lane owns column c = c0+col, rows n contiguous (half*8 + r): fused residual
  const int c = c0 + col;
  const size_t base = ((size_t)b * kC + c) * kN + n0 + half * 8;
#pragma unroll
  for (int r = 0; r < 8; ++r) y[base + r] = vin[base + r] + acc[r];
}

// ---------------------------------------------------------------------------
extern "C" void kernel_launch(void* const* d_in, const int* in_sizes, int n_in,
                              void* d_out, int out_size, void* d_ws, size_t ws_size,
                              hipStream_t stream) {
  (void)in_sizes; (void)n_in; (void)out_size; (void)ws_size;
  const float* q  = (const float*)d_in[0];
  const float* k  = (const float*)d_in[1];
  const float* v  = (const float*)d_in[2];
  const float* Wt = (const float*)d_in[3];
  const float* Wp = (const float*)d_in[4];
  const float* Wg = (const float*)d_in[5];
  const float* Wo = (const float*)d_in[6];
  float* y = (float*)d_out;

  const size_t nd = (size_t)kB * kN * kD;        // elements per bf16 tensor
  unsigned short* qp  = (unsigned short*)d_ws;   // [B][N][D] bf16
  unsigned short* kp  = qp  + nd;                // [B][N][D] bf16
  unsigned short* vpT = kp  + nd;                // [B][D][N] bf16
  unsigned short* ob  = vpT + nd;                // [B][N][D] bf16
  float*          cmx = (float*)(ob + nd);       // [B][N] column max
  float*          cs  = cmx + (size_t)kB * kN;   // [B][N] column exp-sum

  const dim3 blk(128);

  proj_gemm<false><<<dim3(kN / 16, kD / 64, kB), blk, 0, stream>>>(q, Wt, qp);
  proj_gemm<false><<<dim3(kN / 16, kD / 64, kB), blk, 0, stream>>>(k, Wp, kp);
  proj_gemm<true ><<<dim3(kN / 16, kD / 64, kB), blk, 0, stream>>>(v, Wg, vpT);

  col_stats<<<dim3(kN / 16, 1, kB), blk, 0, stream>>>(qp, kp, cmx, cs);

  attn_apply<<<dim3(kN / 16, 1, kB), blk, 0, stream>>>(qp, kp, vpT, cmx, cs, ob);

  out_proj<<<dim3(kN / 16, kC / 64, kB), blk, 0, stream>>>(ob, Wo, v, y);
}
